// When2commFusion_31928786878614
// MI455X (gfx1250) — compile-verified, matrix operand order checked by reading, not run on hardware
//
#include <hip/hip_runtime.h>
#include <hip/hip_bf16.h>
#include <math.h>

typedef _Float16 v8h  __attribute__((ext_vector_type(8)));
typedef _Float16 v16h __attribute__((ext_vector_type(16)));
typedef float    v8f  __attribute__((ext_vector_type(8)));

#define EPSC 1e-5f
// problem dims
#define Bn   4
#define Ln   5
#define NI   20            // B*L
#define Cin  256
#define Hh   100
#define Wd   252
// conv1: 256->512 k4 s4 p0 -> 25x63 (non-overlapping patches)
#define OC1  512
#define OH1  25
#define OW1  63
#define P1   1575          // OH1*OW1
#define K1   4096          // Cin*16
#define M1   31500         // NI*P1
#define MT1  1969          // ceil(M1/16)
// conv2: 512->256 k4 s4 p0 -> 6x15
#define OC2  256
#define OH2  6
#define OW2  15
#define P2   90
#define K2   8192          // OC1*16
#define M2   1800
#define MT2  113
// conv3: 256->256 k3 s2 p1 -> 3x8
#define OC3  256
#define P3   24

// ---------------- workspace layout (bytes) ----------------
#define OFF_WARP  ((size_t)0)                  // f16  NI*P1*K1      = 258,048,000 B
#define OFF_W1H   ((size_t)258048000)          // f16  512*4096      =   4,194,304 B
#define OFF_W2H   ((size_t)262242304)          // f16  256*8192      =   4,194,304 B
#define OFF_OUT1  ((size_t)266436608)          // f16  NI*P2*K2      =  29,491,200 B
#define OFF_OUT2  ((size_t)295927808)          // f32  NI*90*256     =   1,843,200 B
#define OFF_OUT3  ((size_t)297771008)          // f32  NI*24*256     =     491,520 B
#define OFF_POOL  ((size_t)298262528)          // f32  NI*256        =      20,480 B
#define OFF_ATTN  ((size_t)298283008)          // f32  B*L           =          80 B
#define OFF_ST    ((size_t)298283264)          // f32  s1,t1,s2,t2,s3,t3 = 8192 B
#define OFF_W3T   ((size_t)298291456)          // f32  256*9*256     =   2,359,296 B

static __device__ __forceinline__ v16h cat16(v8h lo, v8h hi) {
  return __builtin_shufflevector(lo, hi, 0,1,2,3,4,5,6,7,8,9,10,11,12,13,14,15);
}

// ---------------- 0: weight cast + folded BN scale/bias + conv3 transpose ---------
__global__ void prep_kernel(const float* __restrict__ w1, const float* __restrict__ w2,
    const float* __restrict__ w3,
    const float* b1,const float* g1,const float* be1,const float* mu1,const float* va1,
    const float* b2,const float* g2,const float* be2,const float* mu2,const float* va2,
    const float* b3,const float* g3,const float* be3,const float* mu3,const float* va3,
    _Float16* __restrict__ w1h, _Float16* __restrict__ w2h,
    float* __restrict__ w3t, float* __restrict__ st)
{
  int i = blockIdx.x * 256 + threadIdx.x;
  if (i < OC1 * K1) { w1h[i] = (_Float16)w1[i]; w2h[i] = (_Float16)w2[i]; }
  if (i < OC3 * 2304) {                        // OIHW -> [oc][ky*3+kx][ic]
    int oc = i / 2304, r = i - oc * 2304;
    int ic = r / 9, kk = r - ic * 9;
    w3t[oc * 2304 + kk * 256 + ic] = w3[i];
  }
  if (i < 512) {
    float s = g1[i] * rsqrtf(va1[i] + EPSC);
    st[i] = s; st[512 + i] = (b1[i] - mu1[i]) * s + be1[i];
  }
  if (i < 256) {
    float s2 = g2[i] * rsqrtf(va2[i] + EPSC);
    st[1024 + i] = s2; st[1280 + i] = (b2[i] - mu2[i]) * s2 + be2[i];
    float s3 = g3[i] * rsqrtf(va3[i] + EPSC);
    st[1536 + i] = s3; st[1792 + i] = (b3[i] - mu3[i]) * s3 + be3[i];
  }
}

// ---------------- 1: bilinear warp -> f16 patch layout ----------------
// layout: wh[((n*P1 + p)*K1 + k)], k = c*16 + (y%4)*4 + (x%4), p = (y/4)*63 + x/4
__global__ void warp_kernel(const float* __restrict__ x, const float* __restrict__ mats,
                            _Float16* __restrict__ wh)
{
  int tid = blockIdx.x * 256 + threadIdx.x;          // NI*P1*K1 = 129,024,000
  int k = tid & (K1 - 1);
  int rest = tid >> 12;
  int p = rest % P1;
  int n = rest / P1;
  int c = k >> 4, sub = k & 15;
  int yy = (p / OW1) * 4 + (sub >> 2);
  int xx = (p % OW1) * 4 + (sub & 3);
  int b = n / Ln, l = n % Ln;
  const float* M = mats + (b * (Ln * Ln) + l) * 6;   // normalized_affine_matrix[b,0,l]
  float gx = ((float)xx + 0.5f) * (2.0f / (float)Wd) - 1.0f;
  float gy = ((float)yy + 0.5f) * (2.0f / (float)Hh) - 1.0f;
  float sx = M[0] * gx + M[1] * gy + M[2];
  float sy = M[3] * gx + M[4] * gy + M[5];
  float ix = (sx + 1.0f) * ((float)Wd * 0.5f) - 0.5f;
  float iy = (sy + 1.0f) * ((float)Hh * 0.5f) - 0.5f;
  float x0f = floorf(ix), y0f = floorf(iy);
  float wx = ix - x0f, wy = iy - y0f;
  int x0 = (int)x0f, y0 = (int)y0f;
  const float* plane = x + ((size_t)n * Cin + c) * (Hh * Wd);
  float acc = 0.f;
#pragma unroll
  for (int dy = 0; dy < 2; ++dy) {
    int yi = y0 + dy;
    float wyv = dy ? wy : 1.0f - wy;
    bool yok = (yi >= 0 && yi < Hh);
    int yc = yi < 0 ? 0 : (yi > Hh - 1 ? Hh - 1 : yi);
#pragma unroll
    for (int dx = 0; dx < 2; ++dx) {
      int xi = x0 + dx;
      float wxv = dx ? wx : 1.0f - wx;
      bool ok = yok && (xi >= 0 && xi < Wd);
      int xc = xi < 0 ? 0 : (xi > Wd - 1 ? Wd - 1 : xi);
      float v = plane[yc * Wd + xc];
      acc += ok ? v * wyv * wxv : 0.f;
    }
  }
  wh[tid] = (_Float16)acc;
}

// ---------------- 2: conv1 WMMA GEMM, 2x-unrolled all-loads-first pipeline ---------
__global__ __launch_bounds__(256) void gemm1_kernel(const _Float16* __restrict__ Aw,
    const _Float16* __restrict__ Bw, const float* __restrict__ st,
    _Float16* __restrict__ out1h)
{
  const int lane = threadIdx.x & 31;
  const int wv   = threadIdx.x >> 5;                 // 0..7 -> oc blocks of 64
  const int mtile = blockIdx.x;                      // 0..MT1-1
  const int ocb = wv << 6;
  int row = mtile * 16 + (lane & 15);
  if (row > M1 - 1) row = M1 - 1;
  const int sel = (lane & 16) ? 8 : 0;
  const _Float16* ap = Aw + (size_t)row * K1 + sel;
  const _Float16* bp = Bw + (size_t)(ocb + (lane & 15)) * K1 + sel;
  const size_t js = (size_t)16 * K1;
  v8f acc[4] = {};
  for (int k0 = 0; k0 < K1; k0 += 64) {
    // issue ALL 20 loads for this 64-wide k step before any WMMA:
    v8h a0l = *(const v8h*)(ap + k0);
    v8h a0h = *(const v8h*)(ap + k0 + 16);
    v8h b0l[4], b0h[4], b1l[4], b1h[4];
#pragma unroll
    for (int j = 0; j < 4; ++j) {
      b0l[j] = *(const v8h*)(bp + j * js + k0);
      b0h[j] = *(const v8h*)(bp + j * js + k0 + 16);
    }
    v8h a1l = *(const v8h*)(ap + k0 + 32);
    v8h a1h = *(const v8h*)(ap + k0 + 48);
#pragma unroll
    for (int j = 0; j < 4; ++j) {
      b1l[j] = *(const v8h*)(bp + j * js + k0 + 32);
      b1h[j] = *(const v8h*)(bp + j * js + k0 + 48);
    }
    __builtin_prefetch(ap + k0 + 2048, 0, 1);
    // group 1: only needs first 10 loads -> partial wait, rest stay in flight
    {
      v16h a = cat16(a0l, a0h);
#pragma unroll
      for (int j = 0; j < 4; ++j)
        acc[j] = __builtin_amdgcn_wmma_f32_16x16x32_f16(
            false, a, false, cat16(b0l[j], b0h[j]), (short)0, acc[j], false, false);
    }
    // group 2
    {
      v16h a = cat16(a1l, a1h);
#pragma unroll
      for (int j = 0; j < 4; ++j)
        acc[j] = __builtin_amdgcn_wmma_f32_16x16x32_f16(
            false, a, false, cat16(b1l[j], b1h[j]), (short)0, acc[j], false, false);
    }
  }
  const int ncol  = lane & 15;
  const int mbase = mtile * 16 + ((lane & 16) ? 8 : 0);
#pragma unroll
  for (int j = 0; j < 4; ++j) {
    int oc = ocb + j * 16 + ncol;
    float s = st[oc], t = st[512 + oc];
#pragma unroll
    for (int r = 0; r < 8; ++r) {
      int m = mbase + r;
      if (m >= M1) continue;
      float v = acc[j][r] * s + t;
      v = v > 0.f ? v : 0.f;
      int n = m / P1, p = m - n * P1;
      int oy = p / OW1, ox = p - oy * OW1;
      if (oy < 24 && ox < 60) {                      // region consumed by conv2
        int p2 = (oy >> 2) * OW2 + (ox >> 2);
        int k2 = (oc << 4) + ((oy & 3) << 2) + (ox & 3);
        out1h[((size_t)n * P2 + p2) * K2 + k2] = (_Float16)v;
      }
    }
  }
}

// ---------------- 3: conv2 WMMA GEMM, same pipeline -> f32 NHWC -------------------
__global__ __launch_bounds__(256) void gemm2_kernel(const _Float16* __restrict__ Aw,
    const _Float16* __restrict__ Bw, const float* __restrict__ st2,
    float* __restrict__ out2)
{
  const int lane = threadIdx.x & 31;
  const int wid = blockIdx.x * 8 + (threadIdx.x >> 5);
  if (wid >= MT2 * 4) return;                        // wave-uniform exit
  const int mtile = wid >> 2;
  const int ocb = (wid & 3) << 6;
  int row = mtile * 16 + (lane & 15);
  if (row > M2 - 1) row = M2 - 1;
  const int sel = (lane & 16) ? 8 : 0;
  const _Float16* ap = Aw + (size_t)row * K2 + sel;
  const _Float16* bp = Bw + (size_t)(ocb + (lane & 15)) * K2 + sel;
  const size_t js = (size_t)16 * K2;
  v8f acc[4] = {};
  for (int k0 = 0; k0 < K2; k0 += 64) {
    v8h a0l = *(const v8h*)(ap + k0);
    v8h a0h = *(const v8h*)(ap + k0 + 16);
    v8h b0l[4], b0h[4], b1l[4], b1h[4];
#pragma unroll
    for (int j = 0; j < 4; ++j) {
      b0l[j] = *(const v8h*)(bp + j * js + k0);
      b0h[j] = *(const v8h*)(bp + j * js + k0 + 16);
    }
    v8h a1l = *(const v8h*)(ap + k0 + 32);
    v8h a1h = *(const v8h*)(ap + k0 + 48);
#pragma unroll
    for (int j = 0; j < 4; ++j) {
      b1l[j] = *(const v8h*)(bp + j * js + k0 + 32);
      b1h[j] = *(const v8h*)(bp + j * js + k0 + 48);
    }
    __builtin_prefetch(ap + k0 + 2048, 0, 1);
    {
      v16h a = cat16(a0l, a0h);
#pragma unroll
      for (int j = 0; j < 4; ++j)
        acc[j] = __builtin_amdgcn_wmma_f32_16x16x32_f16(
            false, a, false, cat16(b0l[j], b0h[j]), (short)0, acc[j], false, false);
    }
    {
      v16h a = cat16(a1l, a1h);
#pragma unroll
      for (int j = 0; j < 4; ++j)
        acc[j] = __builtin_amdgcn_wmma_f32_16x16x32_f16(
            false, a, false, cat16(b1l[j], b1h[j]), (short)0, acc[j], false, false);
    }
  }
  const int ncol  = lane & 15;
  const int mbase = mtile * 16 + ((lane & 16) ? 8 : 0);
#pragma unroll
  for (int j = 0; j < 4; ++j) {
    int oc = ocb + j * 16 + ncol;
    float s = st2[oc], t = st2[256 + oc];
#pragma unroll
    for (int r = 0; r < 8; ++r) {
      int m = mbase + r;
      if (m >= M2) continue;
      float v = acc[j][r] * s + t;
      out2[(size_t)m * OC2 + oc] = v > 0.f ? v : 0.f;
    }
  }
}

// ---------------- 4: conv3 (tiny, VALU) -> f32 [n][3*8][256] ----------------
__global__ void conv3_kernel(const float* __restrict__ in2, const float* __restrict__ w3t,
                             const float* __restrict__ st3, float* __restrict__ out3)
{
  int tid = blockIdx.x * 256 + threadIdx.x;          // 20*24*256 = 122880
  if (tid >= NI * P3 * OC3) return;
  int oc = tid & 255;
  int pos = (tid >> 8) % P3;
  int n = (tid >> 8) / P3;
  int oy = pos >> 3, ox = pos & 7;
  float acc = 0.f;
  for (int ky = 0; ky < 3; ++ky) {
    int iy = oy * 2 - 1 + ky;
    if ((unsigned)iy >= (unsigned)OH2) continue;
    for (int kx = 0; kx < 3; ++kx) {
      int ix = ox * 2 - 1 + kx;
      if ((unsigned)ix >= (unsigned)OW2) continue;
      const float* ip = in2 + ((size_t)(n * OH2 + iy) * OW2 + ix) * OC2;
      const float* wp = w3t + (size_t)oc * 2304 + (ky * 3 + kx) * 256;
      for (int ic = 0; ic < 256; ++ic) acc += ip[ic] * wp[ic];
    }
  }
  float v = acc * st3[oc] + st3[256 + oc];
  out3[tid] = v > 0.f ? v : 0.f;
}

// ---------------- 5: spatial mean pool ----------------
__global__ void pool_kernel(const float* __restrict__ out3, float* __restrict__ pooled)
{
  int tid = blockIdx.x * 256 + threadIdx.x;          // NI*256 = 5120
  if (tid >= NI * OC3) return;
  int oc = tid & 255, n = tid >> 8;
  float s = 0.f;
  for (int p = 0; p < P3; ++p) s += out3[((size_t)n * P3 + p) * OC3 + oc];
  pooled[tid] = s * (1.0f / P3);
}

// ---------------- 6: MLPs + attention + softmax (1 block) ----------------
__global__ __launch_bounds__(256) void head_kernel(const float* __restrict__ pooled,
    const float* kw0, const float* kb0, const float* kw1, const float* kb1,
    const float* kw2, const float* kb2,
    const float* qw0, const float* qb0, const float* qw1, const float* qb1,
    const float* qw2, const float* qb2,
    const float* fw, const float* fb, const float* cw, const float* cb,
    const float* ow, const float* obias, float* __restrict__ attnW)
{
  __shared__ float KEYS[NI * 256];
  __shared__ float QRY[Bn * 32];
  __shared__ float A[256], Bf[256], R[256];
  const int t = threadIdx.x;
  // key MLP for all 20 rows
  for (int n = 0; n < NI; ++n) {
    A[t] = pooled[n * 256 + t];
    __syncthreads();
    float v = kb0[t];
    for (int i = 0; i < 256; ++i) v += kw0[t * 256 + i] * A[i];
    Bf[t] = v > 0.f ? v : 0.f;
    __syncthreads();
    if (t < 128) {
      float u = kb1[t];
      for (int i = 0; i < 256; ++i) u += kw1[t * 256 + i] * Bf[i];
      R[t] = u > 0.f ? u : 0.f;
    }
    __syncthreads();
    float w = kb2[t];
    for (int i = 0; i < 128; ++i) w += kw2[t * 128 + i] * R[i];
    KEYS[n * 256 + t] = w;
    __syncthreads();
  }
  // query MLP per batch (l==0 row)
  for (int b = 0; b < Bn; ++b) {
    A[t] = pooled[(b * Ln) * 256 + t];
    __syncthreads();
    float v = qb0[t];
    for (int i = 0; i < 256; ++i) v += qw0[t * 256 + i] * A[i];
    Bf[t] = v > 0.f ? v : 0.f;
    __syncthreads();
    if (t < 128) {
      float u = qb1[t];
      for (int i = 0; i < 256; ++i) u += qw1[t * 256 + i] * Bf[i];
      R[t] = u > 0.f ? u : 0.f;
    }
    __syncthreads();
    if (t < 32) {
      float w = qb2[t];
      for (int i = 0; i < 128; ++i) w += qw2[t * 128 + i] * R[i];
      QRY[b * 32 + t] = w;
    }
    __syncthreads();
  }
  // attention scores + softmax
  for (int b = 0; b < Bn; ++b) {
    float cv = cb[t];
    for (int i = 0; i < 32; ++i) cv += cw[t * 32 + i] * QRY[b * 32 + i];
    float sc[Ln];
    for (int l = 0; l < Ln; ++l) {
      float fv = fb[t];
      const float* kk = &KEYS[(b * Ln + l) * 256];
      for (int i = 0; i < 256; ++i) fv += fw[t * 256 + i] * kk[i];
      R[t] = tanhf(fv + cv) * ow[t];
      __syncthreads();
      for (int s = 128; s > 0; s >>= 1) {
        if (t < s) R[t] += R[t + s];
        __syncthreads();
      }
      sc[l] = R[0] + obias[0];
      __syncthreads();
    }
    if (t == 0) {
      float mx = sc[0];
      for (int l = 1; l < Ln; ++l) mx = fmaxf(mx, sc[l]);
      float sm = 0.f, e[Ln];
      for (int l = 0; l < Ln; ++l) { e[l] = expf(sc[l] - mx); sm += e[l]; }
      for (int l = 0; l < Ln; ++l) attnW[b * Ln + l] = e[l] / sm;
    }
    __syncthreads();
  }
}

// ---------------- 7: attn-weighted fusion, coalesced reads of patch layout ----------
__global__ void fuse_kernel(const _Float16* __restrict__ wh, const float* __restrict__ attnW,
                            float* __restrict__ out)
{
  int tid = blockIdx.x * 256 + threadIdx.x;          // Bn*P1*K1 = 25,804,800
  int k = tid & (K1 - 1);
  int rest = tid >> 12;
  int p = rest % P1;
  int b = rest / P1;
  int c = k >> 4, sub = k & 15;
  int y  = (p / OW1) * 4 + (sub >> 2);
  int xx = (p % OW1) * 4 + (sub & 3);
  float acc = 0.f;
#pragma unroll
  for (int l = 0; l < Ln; ++l)
    acc += attnW[b * Ln + l] * (float)wh[(((size_t)(b * Ln + l)) * P1 + p) * K1 + k];
  out[(((size_t)b * Cin + c) * Hh + y) * Wd + xx] = acc;
}

extern "C" void kernel_launch(void* const* d_in, const int* in_sizes, int n_in,
                              void* d_out, int out_size, void* d_ws, size_t ws_size,
                              hipStream_t stream)
{
  (void)in_sizes; (void)n_in; (void)out_size; (void)ws_size;
  // setup_inputs() insertion order, params flattened in insertion order:
  const float* x    = (const float*)d_in[0];
  // d_in[1] = record_len (unused; always == L)
  const float* mats = (const float*)d_in[2];
  const float* w1   = (const float*)d_in[3];
  const float* b1   = (const float*)d_in[4];
  const float* g1   = (const float*)d_in[5];
  const float* be1  = (const float*)d_in[6];
  const float* mu1  = (const float*)d_in[7];
  const float* va1  = (const float*)d_in[8];
  const float* w2   = (const float*)d_in[9];
  const float* b2   = (const float*)d_in[10];
  const float* g2   = (const float*)d_in[11];
  const float* be2  = (const float*)d_in[12];
  const float* mu2  = (const float*)d_in[13];
  const float* va2  = (const float*)d_in[14];
  const float* w3   = (const float*)d_in[15];
  const float* b3   = (const float*)d_in[16];
  const float* g3   = (const float*)d_in[17];
  const float* be3  = (const float*)d_in[18];
  const float* mu3  = (const float*)d_in[19];
  const float* va3  = (const float*)d_in[20];
  const float* kw0  = (const float*)d_in[21];
  const float* kb0  = (const float*)d_in[22];
  const float* kw1  = (const float*)d_in[23];
  const float* kb1  = (const float*)d_in[24];
  const float* kw2  = (const float*)d_in[25];
  const float* kb2  = (const float*)d_in[26];
  const float* qw0  = (const float*)d_in[27];
  const float* qb0  = (const float*)d_in[28];
  const float* qw1  = (const float*)d_in[29];
  const float* qb1  = (const float*)d_in[30];
  const float* qw2  = (const float*)d_in[31];
  const float* qb2  = (const float*)d_in[32];
  const float* fw   = (const float*)d_in[33];
  const float* fb   = (const float*)d_in[34];
  const float* cw   = (const float*)d_in[35];
  const float* cb   = (const float*)d_in[36];
  const float* ow   = (const float*)d_in[37];
  const float* obias= (const float*)d_in[38];

  char* ws = (char*)d_ws;
  _Float16* warped = (_Float16*)(ws + OFF_WARP);
  _Float16* w1h    = (_Float16*)(ws + OFF_W1H);
  _Float16* w2h    = (_Float16*)(ws + OFF_W2H);
  _Float16* out1h  = (_Float16*)(ws + OFF_OUT1);
  float*    out2   = (float*)(ws + OFF_OUT2);
  float*    out3   = (float*)(ws + OFF_OUT3);
  float*    pooled = (float*)(ws + OFF_POOL);
  float*    attnW  = (float*)(ws + OFF_ATTN);
  float*    st     = (float*)(ws + OFF_ST);
  float*    w3t    = (float*)(ws + OFF_W3T);
  float*    out    = (float*)d_out;

  prep_kernel<<<(OC1 * K1 + 255) / 256, 256, 0, stream>>>(
      w1, w2, w3, b1, g1, be1, mu1, va1, b2, g2, be2, mu2, va2,
      b3, g3, be3, mu3, va3, w1h, w2h, w3t, st);

  warp_kernel<<<(NI * P1 * (K1 / 256)), 256, 0, stream>>>(x, mats, warped);

  gemm1_kernel<<<MT1, 256, 0, stream>>>(warped, w1h, st, out1h);

  gemm2_kernel<<<(MT2 * 4 + 7) / 8, 256, 0, stream>>>(out1h, w2h, st + 1024, out2);

  conv3_kernel<<<(NI * P3 * OC3 + 255) / 256, 256, 0, stream>>>(out2, w3t, st + 1536, out3);

  pool_kernel<<<(NI * OC3 + 255) / 256, 256, 0, stream>>>(out3, pooled);

  head_kernel<<<1, 256, 0, stream>>>(pooled, kw0, kb0, kw1, kb1, kw2, kb2,
                                     qw0, qb0, qw1, qb1, qw2, qb2,
                                     fw, fb, cw, cb, ow, obias, attnW);

  fuse_kernel<<<(Bn * P1 * (K1 / 256)), 256, 0, stream>>>(warped, attnW, out);
}
